// CausalSelfAttention_9483287789860
// MI455X (gfx1250) — matrix operand order repaired and checked by reading, not compiled, not run on hardware
//
#include <hip/hip_runtime.h>
#include <hip/hip_bf16.h>

// ---------------------------------------------------------------------------
// CDNA5 (gfx1250) attention: f16 WMMA GEMMs + flash attention w/ sliding
// window + global tokens.  wave32 everywhere.  Fragment loads use a contiguous-K
// order (k = 16*half + s) so each operand is fetched as 2x b128.  GEMM uses 64x64
// register blocking per wave: 16 WMMAs per 32-K step.
// ---------------------------------------------------------------------------

typedef __attribute__((ext_vector_type(16))) _Float16 v16h;
typedef __attribute__((ext_vector_type(8)))  _Float16 v8h;
typedef __attribute__((ext_vector_type(8)))  float    v8f;

union HFrag  { v16h v; _Float16 h[16]; v8h h8[2]; };

#define T_SEQ   2048
#define C_DIM   2048
#define NHEAD   32
#define NKV     8
#define HD      64
#define KVW     (NKV * HD)   // 512
#define WINDOW  512
#define NGLOB   16

#define WMMA_F16(a, b, c) __builtin_amdgcn_wmma_f32_16x16x32_f16( \
    false, (a), false, (b), (short)0, (c), false, false)

// ---- WMMA fragment loaders (contiguous-K order, 16-bit operands) -------------
// A (16x32): lane 0-15 = row M, holds K 0..15;
//            lane 16-31 same rows, K 16..31.  2x b128.
__device__ __forceinline__ v16h load_fragA_vec(const _Float16* p,
                                               int row_stride, int lane) {
  const _Float16* r =
      p + (size_t)(lane & 15) * row_stride + ((lane & 16) ? 16 : 0);
  HFrag f;
  f.h8[0] = *(const v8h*)(r);
  f.h8[1] = *(const v8h*)(r + 8);
  return f.v;
}
// B (32x16): lane 0-15 = col N, holds K 0..15; lanes 16-31 hold K 16..31.
// One contiguous 16-half run -> 2x b128.
__device__ __forceinline__ v16h load_fragB_vec(const _Float16* p,
                                               int n_stride, int lane) {
  const _Float16* r =
      p + (size_t)(lane & 15) * n_stride + ((lane & 16) ? 16 : 0);
  HFrag f;
  f.h8[0] = *(const v8h*)(r);
  f.h8[1] = *(const v8h*)(r + 8);
  return f.v;
}
// Scalar B loader for K-strided operands (V read out of LDS).
__device__ __forceinline__ v16h load_fragB_str(const _Float16* p, int n_stride,
                                               int k_stride, int lane) {
  HFrag f;
  const int n = lane & 15;
  const int kb = (lane & 16) ? 16 : 0;
#pragma unroll
  for (int e = 0; e < 16; ++e)
    f.h[e] = p[n * n_stride + (kb + e) * k_stride];
  return f.v;
}

// ---- fp32 -> f16 cast ------------------------------------------------------
__global__ void cast_f32_f16_kernel(const float* __restrict__ src,
                                    _Float16* __restrict__ dst, int n) {
  int i = blockIdx.x * blockDim.x + threadIdx.x;
  if (i < n) dst[i] = (_Float16)src[i];
}

// ---- C = A(MxK) * B^T(NxK) via v_wmma_f32_16x16x32_f16 --------------------
// One wave computes a 64x64 tile (4x4 WMMA subtiles => 16 WMMAs per K-step,
// 16 b128 loads per K-step -> 1.0 wmma per vmem instruction).
__global__ void gemm_f16_wmma_kernel(const _Float16* __restrict__ A,
                                     const _Float16* __restrict__ Bw,
                                     float* __restrict__ C,
                                     int M, int N, int K) {
  const int lane = threadIdx.x & 31;
  const int wave = threadIdx.x >> 5;
  const int tile = blockIdx.x * (blockDim.x >> 5) + wave;
  const int n64 = N >> 6;
  if (tile >= (M >> 6) * n64) return;
  const int m0 = (tile / n64) << 6;
  const int n0 = (tile % n64) << 6;

  __builtin_amdgcn_s_wait_tensorcnt(0);   // gfx1250 split-counter touchpoint

  v8f acc[4][4];
#pragma unroll
  for (int sm = 0; sm < 4; ++sm)
#pragma unroll
    for (int sn = 0; sn < 4; ++sn)
#pragma unroll
      for (int i = 0; i < 8; ++i) acc[sm][sn][i] = 0.0f;

  const _Float16* ap = A + (size_t)m0 * K;
  const _Float16* bp = Bw + (size_t)n0 * K;

  for (int k0 = 0; k0 < K; k0 += 32) {
    if (k0 + 32 < K) {                      // prefetch next K-tile
      __builtin_prefetch(ap + k0 + 32, 0, 3);
      __builtin_prefetch(bp + k0 + 32, 0, 3);
    }
    v16h bfr[4];
#pragma unroll
    for (int sn = 0; sn < 4; ++sn)
      bfr[sn] = load_fragB_vec(bp + (size_t)(sn * 16) * K + k0, K, lane);
#pragma unroll
    for (int sm = 0; sm < 4; ++sm) {
      v16h afr = load_fragA_vec(ap + (size_t)(sm * 16) * K + k0, K, lane);
#pragma unroll
      for (int sn = 0; sn < 4; ++sn)
        acc[sm][sn] = WMMA_F16(afr, bfr[sn], acc[sm][sn]);
    }
  }

  // C/D layout: VGPR i -> row (i or i+8 per half-wave), col = lane&15
  const int col = n0 + (lane & 15);
  const int rb = m0 + ((lane >> 4) << 3);
#pragma unroll
  for (int sm = 0; sm < 4; ++sm)
#pragma unroll
    for (int sn = 0; sn < 4; ++sn)
#pragma unroll
      for (int i = 0; i < 8; ++i)
        C[(size_t)(rb + sm * 16 + i) * N + col + sn * 16] = acc[sm][sn][i];
}

// ---- RoPE (interleaved pairs) + cast to f16 --------------------------------
__global__ void rope_cast_kernel(const float* __restrict__ src,
                                 _Float16* __restrict__ dst,
                                 const float* __restrict__ fc,
                                 const float* __restrict__ fs,
                                 int rowwidth) {
  const int pairs_per_row = rowwidth >> 1;
  int idx = blockIdx.x * blockDim.x + threadIdx.x;
  if (idx >= T_SEQ * pairs_per_row) return;
  const int t = idx / pairs_per_row;
  const int pr = idx % pairs_per_row;
  const int j = pr & 31;                    // pair index within head (HD/2=32)
  const int base = t * rowwidth + ((pr >> 5) << 6) + (j << 1);
  const float x0 = src[base], x1 = src[base + 1];
  const float c = fc[t * 32 + j], s = fs[t * 32 + j];
  dst[base]     = (_Float16)(x0 * c - x1 * s);
  dst[base + 1] = (_Float16)(x0 * s + x1 * c);
}

// ---- flash attention: one wave per (head, 16-query tile) -------------------
__global__ void attn_flash_kernel(const _Float16* __restrict__ q,  // T x C
                                  const _Float16* __restrict__ k,  // T x 512
                                  const _Float16* __restrict__ v,  // T x 512
                                  float* __restrict__ y) {         // T x C
  __shared__ __align__(16) _Float16 pbuf[16 * 32];   // P tile (1 KB)
  __shared__ __align__(16) _Float16 vbuf[32 * 64];   // V tile [kv][hd] (4 KB)
  const int lane = threadIdx.x & 31;
  const int h  = blockIdx.x >> 7;          // / (T/16)
  const int qt = blockIdx.x & 127;
  const int q0 = qt << 4;
  const int g  = h >> 2;                   // kv head (N_REP = 4)
  const float scale = 0.125f;              // 1/sqrt(64)

  // Q fragments: 16 rows x 64 (two K-halves of 32), b128 loads
  const _Float16* qb = q + (size_t)q0 * C_DIM + h * HD;
  const v16h qa0 = load_fragA_vec(qb,      C_DIM, lane);
  const v16h qa1 = load_fragA_vec(qb + 32, C_DIM, lane);

  v8f o[4];
  float mrow[8], lrow[8];
#pragma unroll
  for (int t = 0; t < 4; ++t)
#pragma unroll
    for (int i = 0; i < 8; ++i) o[t][i] = 0.0f;
#pragma unroll
  for (int i = 0; i < 8; ++i) { mrow[i] = -1e30f; lrow[i] = 0.0f; }

  int wstart = q0 - (WINDOW - 1);
  if (wstart < 0) wstart = 0;
  const int kv_lo = wstart & ~31;
  const int extra = (kv_lo > 0) ? 1 : 0;   // separate block for global tokens
  const int nblk = ((q0 + 16) - kv_lo + 31) >> 5;

  for (int b = 0; b < nblk + extra; ++b) {
    const int kv0 = (b < extra) ? 0 : kv_lo + ((b - extra) << 5);

    // stage V tile into LDS: lane = kv row, coalesced 8x b128 per lane
    {
      const _Float16* vrow = v + (size_t)(kv0 + lane) * KVW + g * HD;
#pragma unroll
      for (int c = 0; c < 8; ++c)
        *(v8h*)(vbuf + lane * 64 + c * 8) = *(const v8h*)(vrow + c * 8);
    }

    // S = Q * K^T for 32 key columns (4 WMMAs); K^T fragments are b128 loads
    v8f s0, s1;
#pragma unroll
    for (int i = 0; i < 8; ++i) { s0[i] = 0.0f; s1[i] = 0.0f; }
    const _Float16* kb = k + (size_t)kv0 * KVW + g * HD;
    {
      v16h b00 = load_fragB_vec(kb,                 KVW, lane);
      v16h b01 = load_fragB_vec(kb + 32,            KVW, lane);
      v16h b10 = load_fragB_vec(kb + 16 * KVW,      KVW, lane);
      v16h b11 = load_fragB_vec(kb + 16 * KVW + 32, KVW, lane);
      s0 = WMMA_F16(qa0, b00, s0);
      s0 = WMMA_F16(qa1, b01, s0);
      s1 = WMMA_F16(qa0, b10, s1);
      s1 = WMMA_F16(qa1, b11, s1);
    }

    // mask (sliding window + global tokens) and per-row max
    const int col_lo = kv0 + (lane & 15);
    const int col_hi = col_lo + 16;
    float cmax[8];
#pragma unroll
    for (int i = 0; i < 8; ++i) {
      const int row = q0 + i + ((lane >> 4) << 3);
      const bool v0 = ((col_lo <= row) && (col_lo >= row - (WINDOW - 1))) || (col_lo < NGLOB);
      const bool v1 = ((col_hi <= row) && (col_hi >= row - (WINDOW - 1))) || (col_hi < NGLOB);
      const float a0 = v0 ? s0[i] * scale : -3e38f;
      const float a1 = v1 ? s1[i] * scale : -3e38f;
      s0[i] = a0; s1[i] = a1;
      cmax[i] = fmaxf(a0, a1);
    }
#pragma unroll
    for (int off = 1; off < 16; off <<= 1)
#pragma unroll
      for (int i = 0; i < 8; ++i)
        cmax[i] = fmaxf(cmax[i], __shfl_xor(cmax[i], off, 32));

    float alpha[8], csum[8];
#pragma unroll
    for (int i = 0; i < 8; ++i) {
      const float mn = fmaxf(mrow[i], cmax[i]);
      alpha[i] = __expf(mrow[i] - mn);
      mrow[i] = mn;
      const float p0 = __expf(s0[i] - mn);
      const float p1 = __expf(s1[i] - mn);
      s0[i] = p0; s1[i] = p1;
      csum[i] = p0 + p1;
    }
#pragma unroll
    for (int off = 1; off < 16; off <<= 1)
#pragma unroll
      for (int i = 0; i < 8; ++i)
        csum[i] += __shfl_xor(csum[i], off, 32);
#pragma unroll
    for (int i = 0; i < 8; ++i) lrow[i] = lrow[i] * alpha[i] + csum[i];
#pragma unroll
    for (int t = 0; t < 4; ++t)
#pragma unroll
      for (int i = 0; i < 8; ++i) o[t][i] *= alpha[i];

    // stage P through LDS (C-fragment -> A-fragment re-layout)
#pragma unroll
    for (int i = 0; i < 8; ++i) {
      const int row = i + ((lane >> 4) << 3);
      pbuf[row * 32 + (lane & 15)]      = (_Float16)s0[i];
      pbuf[row * 32 + (lane & 15) + 16] = (_Float16)s1[i];
    }
    __syncthreads();
    const v16h pa = load_fragA_vec(pbuf, 32, lane);  // ds_load_b128 x2

    // O += P (16x32) * V (32x64): V fragments read from LDS
#pragma unroll
    for (int t = 0; t < 4; ++t) {
      v16h bv = load_fragB_str(vbuf + t * 16, 1, 64, lane);
      o[t] = WMMA_F16(pa, bv, o[t]);
    }
    __syncthreads();
  }

  // normalize and store
#pragma unroll
  for (int i = 0; i < 8; ++i) lrow[i] = 1.0f / lrow[i];
#pragma unroll
  for (int t = 0; t < 4; ++t) {
    const int col = h * HD + (t << 4) + (lane & 15);
#pragma unroll
    for (int i = 0; i < 8; ++i) {
      const int row = q0 + i + ((lane >> 4) << 3);
      y[(size_t)row * C_DIM + col] = o[t][i] * lrow[i];
    }
  }
}

// ---------------------------------------------------------------------------
extern "C" void kernel_launch(void* const* d_in, const int* in_sizes, int n_in,
                              void* d_out, int out_size, void* d_ws, size_t ws_size,
                              hipStream_t stream) {
  const float* x  = (const float*)d_in[0];
  const float* fc = (const float*)d_in[1];
  const float* fs = (const float*)d_in[2];
  const float* wq = (const float*)d_in[3];
  const float* wk = (const float*)d_in[4];
  const float* wv = (const float*)d_in[5];
  const float* wo = (const float*)d_in[6];
  float* out = (float*)d_out;

  const size_t MB = 1024ull * 1024ull;
  char* ws = (char*)d_ws;
  _Float16* x_h  = (_Float16*)(ws + 0 * MB);   // 2048x2048   (8 MB)
  _Float16* wq_h = (_Float16*)(ws + 8 * MB);   // 2048x2048   (8 MB)
  _Float16* wk_h = (_Float16*)(ws + 16 * MB);  // 512x2048    (2 MB)
  _Float16* wv_h = (_Float16*)(ws + 18 * MB);  // 512x2048    (2 MB)
  _Float16* wo_h = (_Float16*)(ws + 20 * MB);  // 2048x2048   (8 MB)
  float*    qf   = (float*)   (ws + 28 * MB);  // 2048x2048   (16 MB)
  float*    kf   = (float*)   (ws + 44 * MB);  // 2048x512    (4 MB)
  float*    vf   = (float*)   (ws + 48 * MB);  // 2048x512    (4 MB)
  _Float16* q_h  = (_Float16*)(ws + 52 * MB);  // 2048x2048   (8 MB)
  _Float16* k_h  = (_Float16*)(ws + 60 * MB);  // 2048x512    (2 MB)
  _Float16* v_h  = (_Float16*)(ws + 62 * MB);  // 2048x512    (2 MB)
  float*    yf   = (float*)   (ws + 64 * MB);  // 2048x2048   (16 MB)
  _Float16* y_h  = (_Float16*)(ws + 80 * MB);  // 2048x2048   (8 MB)

  const int nTC = T_SEQ * C_DIM;   // 4M
  const int nTK = T_SEQ * KVW;     // 1M

  // 1) cast inputs/weights to f16
  cast_f32_f16_kernel<<<(nTC + 255) / 256, 256, 0, stream>>>(x,  x_h,  nTC);
  cast_f32_f16_kernel<<<(nTC + 255) / 256, 256, 0, stream>>>(wq, wq_h, nTC);
  cast_f32_f16_kernel<<<(nTK + 255) / 256, 256, 0, stream>>>(wk, wk_h, nTK);
  cast_f32_f16_kernel<<<(nTK + 255) / 256, 256, 0, stream>>>(wv, wv_h, nTK);
  cast_f32_f16_kernel<<<(nTC + 255) / 256, 256, 0, stream>>>(wo, wo_h, nTC);

  // 2) projections (WMMA GEMMs, 64x64 tile per wave): out = in * W^T
  {
    int tiles = (T_SEQ / 64) * (C_DIM / 64);   // 1024 waves -> 128 blocks
    gemm_f16_wmma_kernel<<<tiles / 8, 256, 0, stream>>>(x_h, wq_h, qf,
                                                        T_SEQ, C_DIM, C_DIM);
    tiles = (T_SEQ / 64) * (KVW / 64);         // 256 waves -> 32 blocks
    gemm_f16_wmma_kernel<<<tiles / 8, 256, 0, stream>>>(x_h, wk_h, kf,
                                                        T_SEQ, KVW, C_DIM);
    gemm_f16_wmma_kernel<<<tiles / 8, 256, 0, stream>>>(x_h, wv_h, vf,
                                                        T_SEQ, KVW, C_DIM);
  }

  // 3) RoPE(+cast) for q/k; plain cast for v
  {
    int nq = T_SEQ * (C_DIM / 2);
    rope_cast_kernel<<<(nq + 255) / 256, 256, 0, stream>>>(qf, q_h, fc, fs, C_DIM);
    int nk = T_SEQ * (KVW / 2);
    rope_cast_kernel<<<(nk + 255) / 256, 256, 0, stream>>>(kf, k_h, fc, fs, KVW);
    cast_f32_f16_kernel<<<(nTK + 255) / 256, 256, 0, stream>>>(vf, v_h, nTK);
  }

  // 4) flash attention: one wave per (head, 16-query tile)
  attn_flash_kernel<<<NHEAD * (T_SEQ / 16), 32, 0, stream>>>(q_h, k_h, v_h, yf);

  // 5) cast attention output, 6) output projection -> d_out
  cast_f32_f16_kernel<<<(nTC + 255) / 256, 256, 0, stream>>>(yf, y_h, nTC);
  {
    int tiles = (T_SEQ / 64) * (C_DIM / 64);
    gemm_f16_wmma_kernel<<<tiles / 8, 256, 0, stream>>>(y_h, wo_h, out,
                                                        T_SEQ, C_DIM, C_DIM);
  }
}
